// GraphLearner_50611894616232
// MI455X (gfx1250) — matrix operand order repaired and checked by reading, not compiled, not run on hardware
//
#include <hip/hip_runtime.h>
#include <hip/hip_bf16.h>
#include <stdint.h>

// Problem constants (match reference)
#define B_   2
#define T_   512
#define N_   512
#define H_   64
#define S_   8
#define C_   8

typedef _Float16 f16;
typedef __attribute__((ext_vector_type(16))) _Float16     v16h;
typedef __attribute__((ext_vector_type(8)))  float        v8f;
typedef __attribute__((ext_vector_type(4)))  unsigned int u32x4;
typedef __attribute__((ext_vector_type(8)))  int          i32x8;
typedef __attribute__((ext_vector_type(4)))  int          i32x4;

// ---- workspace layout (bytes, 256-aligned) ----
#define OFF_DYN   0u          // B*N*4  f32  = 16384
#define OFF_U     16384u      // B*N*64 f32  = 262144
#define OFF_V     278528u     // B*N*64 f32  = 262144
#define OFF_DEG   540672u     // B*N    f32  = 4096
#define OFF_HREP  544768u     // B*N*64 f16  = 131072
#define OFF_BPK   675840u     // 16*32*16 f16 = 16384  (packed W3c/W3d B-fragments)

// ------------------------------------------------------------------
// TDM: 1-D contiguous global->LDS tile load (data_size = 8B units).
// Descriptor per CDNA5 ISA D# groups 0-3; clang-23 6-arg builtin form
// (extra int32x8 operand zero-filled).
// ------------------------------------------------------------------
__device__ __forceinline__ unsigned lds_off(const void* p) {
  // flat LDS address: low 32 bits are the LDS byte offset (ISA aperture rule)
  return (unsigned)(uintptr_t)p;
}

__device__ __forceinline__ void tdm_load_1d(unsigned lds_byte_off,
                                            const void* gsrc,
                                            unsigned nbytes) {
  uint64_t ga = (uint64_t)(uintptr_t)gsrc;
  unsigned n8 = nbytes >> 3;                       // elements of 8 bytes
  u32x4 g0 = { 1u,                                  // count=1, user mode
               lds_byte_off,                        // lds_addr
               (unsigned)(ga & 0xFFFFFFFFu),        // global_addr lo
               (unsigned)((ga >> 32) & 0x01FFFFFFu) | (2u << 30) }; // hi + type=2
  i32x8 g1 = { (int)(3u << 16),                     // data_size = 3 (8B)
               (int)((n8 & 0xFFFFu) << 16),         // tensor_dim0[15:0]
               (int)((n8 >> 16) | (1u << 16)),      // tensor_dim0[31:16] | tensor_dim1=1
               (int)((n8 & 0xFFFFu) << 16),         // tile_dim0
               0,                                   // tile_dim1=0, tile_dim2=0 (1-D)
               (int)n8,                             // tensor_dim0_stride lo32
               0, 0 };
  i32x4 g2 = { 0, 0, 0, 0 };
  i32x4 g3 = { 0, 0, 0, 0 };
  i32x8 g4 = { 0, 0, 0, 0, 0, 0, 0, 0 };
  __builtin_amdgcn_tensor_load_to_lds(g0, g1, g2, g3, g4, 0);
}

// ------------------------------------------------------------------
// Kernel P: pack W3c/W3d (rows 128..255 of W3) into WMMA B-fragment
// layout: frag f = ((mat*2+kc)*4 + ntile); lane l, slot s:
//   k = kc*32 + (l<16?0:16) + s ; n = ntile*16 + (l&15)
// ------------------------------------------------------------------
__global__ void k_pack(const float* __restrict__ W3, f16* __restrict__ Bpack) {
  for (int e = threadIdx.x; e < 16 * 32 * 16; e += 256) {
    int f    = e >> 9;
    int lane = (e >> 4) & 31;
    int slot = e & 15;
    int mat  = f >> 3;
    int kc   = (f >> 2) & 1;
    int t    = f & 3;
    int krow = 128 + mat * 64 + kc * 32 + ((lane & 16) ? 16 : 0) + slot;
    int col  = t * 16 + (lane & 15);
    Bpack[e] = (f16)W3[krow * H_ + col];
  }
}

// ------------------------------------------------------------------
// Kernel A: dynamic context (mean/std/last/missing_ratio). One thread
// per (b,n); lanes over consecutive n -> fully coalesced over T.
// ------------------------------------------------------------------
__global__ void k_dyn(const float* __restrict__ x, const float* __restrict__ mask,
                      float* __restrict__ dyn) {
  int g = blockIdx.x * blockDim.x + threadIdx.x;   // 0 .. B*N-1
  int b = g / N_, n = g % N_;
  const float* xb = x    + (size_t)b * T_ * N_ + n;
  const float* mb = mask + (size_t)b * T_ * N_ + n;
  float sum = 0.f, sumsq = 0.f, cnt = 0.f, msum = 0.f, lastX = 0.f, x0 = 0.f;
  int lastT = -1;
  for (int t = 0; t < T_; ++t) {
    float xv = xb[(size_t)t * N_];
    float mv = mb[(size_t)t * N_];
    float obs = 1.0f - mv;
    if (t == 0) x0 = xv;
    sum   += xv * obs;
    sumsq += xv * xv * obs;
    cnt   += obs;
    msum  += mv;
    if (obs > 0.5f) { lastT = t; lastX = xv; }
  }
  if (lastT < 0) lastX = x0;                        // argmax of all-zeros -> t=0
  float count = fmaxf(cnt, 1.0f);
  float mean  = sum / count;
  float num   = sumsq - 2.0f * mean * sum + mean * mean * cnt;
  float sd    = sqrtf(num / count + 1e-6f);
  dyn[g * 4 + 0] = mean;
  dyn[g * 4 + 1] = sd;
  dyn[g * 4 + 2] = lastX;
  dyn[g * 4 + 3] = msum / (float)T_;
}

// ------------------------------------------------------------------
// Kernel B: node MLP (feats->h1->node_repr), plus U = h@W3a,
// V = h@W3b + b3, and f16 copy of node_repr. One 64-thread block/row.
// ------------------------------------------------------------------
__global__ void k_node(const float* __restrict__ dyn, const float* __restrict__ sctx,
                       const float* __restrict__ W1, const float* __restrict__ b1,
                       const float* __restrict__ W2, const float* __restrict__ b2,
                       const float* __restrict__ W3, const float* __restrict__ b3,
                       float* __restrict__ U, float* __restrict__ V,
                       f16* __restrict__ hrep) {
  int row = blockIdx.x;          // b*N + n
  int n   = row % N_;
  int k   = threadIdx.x;         // 0..63
  __shared__ float sh1[H_];
  __shared__ float sh2[H_];
  float f[12];
#pragma unroll
  for (int i = 0; i < 4; ++i) f[i] = dyn[row * 4 + i];
#pragma unroll
  for (int i = 0; i < 8; ++i) f[4 + i] = sctx[n * 8 + i];
  float h1 = b1[k];
#pragma unroll
  for (int i = 0; i < 12; ++i) h1 += f[i] * W1[i * H_ + k];
  h1 = fmaxf(h1, 0.0f);
  sh1[k] = h1;
  __syncthreads();
  float h2 = b2[k];
  for (int m = 0; m < H_; ++m) h2 += sh1[m] * W2[m * H_ + k];
  h2 = fmaxf(h2, 0.0f);
  sh2[k] = h2;
  hrep[(size_t)row * H_ + k] = (f16)h2;
  __syncthreads();
  float u = 0.0f, v = b3[k];
  for (int m = 0; m < H_; ++m) {
    float hm = sh2[m];
    u += hm * W3[m * H_ + k];              // W3a rows 0..63
    v += hm * W3[(H_ + m) * H_ + k];       // W3b rows 64..127
  }
  U[(size_t)row * H_ + k] = u;
  V[(size_t)row * H_ + k] = v;
}

// ------------------------------------------------------------------
// Kernel C: pairwise edge MLP on WMMA. Block = 256 thr (8 waves),
// handles (b, 16-i tile, 128-j supertile). TDM stages all tiles
// into LDS. Per wave-task (i, 16-j tile): build A = |hi-hj| and
// hi*hj fragments, 16x v_wmma_f32_16x16x32_f16, epilogue writes
// adapt = relu(w4.relu(e+U+V)+b4) * prior + eye.
// ------------------------------------------------------------------
struct SmemC {
  f16   hi[16][H_];
  f16   hj[128][H_];
  float Ui[16][H_];
  float Vj[128][H_];
  float ci[16][C_];
  float cj[128][C_];
  float w4[H_];
  float b4v;
};

__global__ void __launch_bounds__(256) k_pair(
    const f16* __restrict__ hrep, const float* __restrict__ U,
    const float* __restrict__ V, const f16* __restrict__ Bpack,
    const float* __restrict__ coords, const float* __restrict__ W4,
    const float* __restrict__ b4p, float* __restrict__ out) {
  __shared__ SmemC sm;
  int bx  = blockIdx.x;
  int b   = bx >> 7;             // / (32 i-tiles * 4 j-supertiles)
  int rem = bx & 127;
  int it  = rem >> 2;
  int js  = rem & 3;
  int i0 = it * 16, j0 = js * 128;
  int tid = threadIdx.x, wid = tid >> 5, lane = tid & 31;

  if (tid < H_) sm.w4[tid] = W4[tid];
  if (tid == H_) sm.b4v = b4p[0];

  if (wid == 0) {
    tdm_load_1d(lds_off(&sm.hi[0][0]), hrep + (size_t)(b * N_ + i0) * H_, 16 * H_ * 2);
    tdm_load_1d(lds_off(&sm.hj[0][0]), hrep + (size_t)(b * N_ + j0) * H_, 128 * H_ * 2);
    tdm_load_1d(lds_off(&sm.Ui[0][0]), U + (size_t)(b * N_ + i0) * H_, 16 * H_ * 4);
    tdm_load_1d(lds_off(&sm.Vj[0][0]), V + (size_t)(b * N_ + j0) * H_, 128 * H_ * 4);
    tdm_load_1d(lds_off(&sm.ci[0][0]), coords + (size_t)i0 * C_, 16 * C_ * 4);
    tdm_load_1d(lds_off(&sm.cj[0][0]), coords + (size_t)j0 * C_, 128 * C_ * 4);
    __builtin_amdgcn_s_wait_tensorcnt(0);
  }
  __syncthreads();

  // Per-wave resident B fragments: W3c/W3d, 2 k-chunks x 4 n-tiles.
  v16h bfr[16];
#pragma unroll
  for (int f = 0; f < 16; ++f)
    bfr[f] = *(const v16h*)(Bpack + f * 512 + lane * 16);

  int nlo   = lane & 15;
  int jhalf = (lane & 16) ? 8 : 0;
  int koffA = (lane & 16) ? 16 : 0;    // contiguous K split: lanes 16..31 -> K+16

  for (int tt = wid; tt < 128; tt += 8) {
    int il = tt >> 3;      // i within tile
    int jt = tt & 7;       // j-tile within supertile
    v8f acc[4];
#pragma unroll
    for (int t = 0; t < 4; ++t) acc[t] = (v8f){0.f,0.f,0.f,0.f,0.f,0.f,0.f,0.f};

#pragma unroll
    for (int kc = 0; kc < 2; ++kc) {
      v16h aab, amu;
      int kb = kc * 32 + koffA;
#pragma unroll
      for (int s = 0; s < 16; ++s) {
        int k  = kb + s;                     // slots 0..15 -> K contiguous
        float ai = (float)sm.hi[il][k];
        float aj = (float)sm.hj[jt * 16 + nlo][k];   // A row M = j_local = lane%16
        aab[s] = (f16)fabsf(ai - aj);
        amu[s] = (f16)(ai * aj);
      }
#pragma unroll
      for (int t = 0; t < 4; ++t) {
        acc[t] = __builtin_amdgcn_wmma_f32_16x16x32_f16(
            false, aab, false, bfr[kc * 4 + t], (short)0, acc[t], false, false);
        acc[t] = __builtin_amdgcn_wmma_f32_16x16x32_f16(
            false, amu, false, bfr[8 + kc * 4 + t], (short)0, acc[t], false, false);
      }
    }

    // Epilogue: e[j,n] = acc + U_i[n] + V_j[n]; relu; dot w4; relu(+b4);
    // scale by prior(i,j), self-loop on diagonal.
    int i = i0 + il;
#pragma unroll
    for (int r = 0; r < 8; ++r) {
      int jl = r + jhalf;                 // C/D layout: M = r + 8*(lane>=16)
      int jr = jt * 16 + jl;
      float val = 0.0f;
#pragma unroll
      for (int t = 0; t < 4; ++t) {
        int n = nlo + 16 * t;
        float e = acc[t][r] + sm.Ui[il][n] + sm.Vj[jr][n];
        e = fmaxf(e, 0.0f);
        val += sm.w4[n] * e;
      }
      // reduce the 16 n-lanes within each half-wave
      val += __shfl_xor(val, 1, 32);
      val += __shfl_xor(val, 2, 32);
      val += __shfl_xor(val, 4, 32);
      val += __shfl_xor(val, 8, 32);
      if (nlo == 0) {
        float edge = fmaxf(val + sm.b4v, 0.0f);
        int jg = j0 + jr;
        float adapt;
        if (jg == i) {
          adapt = 1.0f;                   // SELF_LOOP_W; prior diag is 0
        } else {
          float d2 = 0.0f;
#pragma unroll
          for (int c = 0; c < C_; ++c) {
            float d = sm.ci[il][c] - sm.cj[jr][c];
            d2 += d * d;
          }
          float dist = (d2 > 0.0f) ? sqrtf(d2) : 0.0f;
          adapt = edge * (1.0f / (1.0f + dist));
        }
        out[(size_t)(b * N_ + i) * N_ + jg] = adapt;
      }
    }
  }
}

// ------------------------------------------------------------------
// Kernel E: deterministic row-sum (degree) -- no float atomics so
// graph replays are bit-stable.
// ------------------------------------------------------------------
__global__ void k_degree(const float* __restrict__ out, float* __restrict__ degree) {
  __shared__ float red[256];
  int row = blockIdx.x;                       // b*N + i
  const float* p = out + (size_t)row * N_;
  float s = p[threadIdx.x] + p[threadIdx.x + 256];
  red[threadIdx.x] = s;
  __syncthreads();
  for (int o = 128; o > 0; o >>= 1) {
    if (threadIdx.x < o) red[threadIdx.x] += red[threadIdx.x + o];
    __syncthreads();
  }
  if (threadIdx.x == 0) degree[row] = red[0];
}

// ------------------------------------------------------------------
// Kernel D: out = D^-1/2 * adapt * D^-1/2 (in place on d_out).
// ------------------------------------------------------------------
__global__ void k_norm(float* __restrict__ out, const float* __restrict__ degree) {
  int g   = blockIdx.x * blockDim.x + threadIdx.x;   // < B*N*N
  int b   = g >> 18;
  int rem = g & (N_ * N_ - 1);
  int i = rem >> 9, j = rem & (N_ - 1);
  float di = fmaxf(degree[b * N_ + i], 1e-6f);
  float dj = fmaxf(degree[b * N_ + j], 1e-6f);
  out[g] = out[g] * (1.0f / sqrtf(di)) * (1.0f / sqrtf(dj));
}

extern "C" void kernel_launch(void* const* d_in, const int* in_sizes, int n_in,
                              void* d_out, int out_size, void* d_ws, size_t ws_size,
                              hipStream_t stream) {
  const float* x      = (const float*)d_in[0];
  const float* mask   = (const float*)d_in[1];
  const float* sctx   = (const float*)d_in[2];
  const float* coords = (const float*)d_in[3];
  const float* W1 = (const float*)d_in[4];
  const float* b1 = (const float*)d_in[5];
  const float* W2 = (const float*)d_in[6];
  const float* b2 = (const float*)d_in[7];
  const float* W3 = (const float*)d_in[8];
  const float* b3 = (const float*)d_in[9];
  const float* W4 = (const float*)d_in[10];
  const float* b4 = (const float*)d_in[11];

  char*  ws     = (char*)d_ws;
  float* dyn    = (float*)(ws + OFF_DYN);
  float* U      = (float*)(ws + OFF_U);
  float* V      = (float*)(ws + OFF_V);
  float* degree = (float*)(ws + OFF_DEG);
  f16*   hrep   = (f16*)(ws + OFF_HREP);
  f16*   Bpack  = (f16*)(ws + OFF_BPK);
  float* out    = (float*)d_out;

  k_pack  <<<1, 256, 0, stream>>>(W3, Bpack);
  k_dyn   <<<(B_ * N_) / 256, 256, 0, stream>>>(x, mask, dyn);
  k_node  <<<B_ * N_, 64, 0, stream>>>(dyn, sctx, W1, b1, W2, b2, W3, b3, U, V, hrep);
  k_pair  <<<B_ * (N_ / 16) * (N_ / 128), 256, 0, stream>>>(hrep, U, V, Bpack,
                                                            coords, W4, b4, out);
  k_degree<<<B_ * N_, 256, 0, stream>>>(out, degree);
  k_norm  <<<(B_ * N_ * N_) / 256, 256, 0, stream>>>(out, degree);
}
